// GeoAttention_7584912245505
// MI455X (gfx1250) — matrix operand order repaired and checked by reading, not compiled, not run on hardware
//
#include <hip/hip_runtime.h>

// ---------------------------------------------------------------------------
// GeoAttention for MI455X (gfx1250), wave32, WMMA f16->f32.
// B=4, N=1024, D=512, H=8, DK=64.
// Round 2: all hot-loop fragments are contiguous f16 (pre-converted /
// pre-transposed), V kept in [B,H,DK,N] so P*V B-frags are unit-stride,
// global_prefetch_b8 on the next key block.
// ---------------------------------------------------------------------------

typedef __attribute__((ext_vector_type(16))) _Float16 v16h;
typedef __attribute__((ext_vector_type(8)))  float    v8f;

#define BATCH   4
#define NSEQ    1024
#define DMODEL  512
#define NHEAD   8
#define DK      64

__device__ __forceinline__ v8f zero8() {
  v8f z;
#pragma unroll
  for (int i = 0; i < 8; ++i) z[i] = 0.0f;
  return z;
}

__device__ __forceinline__ v8f wmma_f16(v16h a, v16h b, v8f c) {
  // D = A(16x32 f16) * B(32x16 f16) + C(16x16 f32)
  return __builtin_amdgcn_wmma_f32_16x16x32_f16(
      /*neg_a=*/false, a, /*neg_b=*/false, b,
      /*c_mod=*/(short)0, c, /*reuse_a=*/false, /*reuse_b=*/false);
}

// A-fragment 16x32 f16: lane&15 = row; lane half selects K {0..15} vs
// {16..31}. p points at (row, k0); one 32B contiguous run.
__device__ __forceinline__ v16h load_a_f16(const _Float16* p, int lane) {
  const int kb = ((lane >> 4) & 1) * 16;
  v16h a;
#pragma unroll
  for (int j = 0; j < 8; ++j) {
    a[j]     = p[kb + j];
    a[8 + j] = p[kb + 8 + j];
  }
  return a;
}

// B-fragment 32x16 f16: lane&15 = column; lanes 0-15 hold K=0..15, lanes
// 16-31 hold K=16..31. Caller pre-offsets p to (k0+ko, col); es = K stride.
__device__ __forceinline__ v16h load_b_f16(const _Float16* p, int es) {
  v16h b;
#pragma unroll
  for (int j = 0; j < 16; ++j) b[j] = p[j * es];
  return b;
}

__device__ __forceinline__ v16h load_b_f32(const float* p, int es) {
  v16h b;
#pragma unroll
  for (int j = 0; j < 16; ++j) b[j] = (_Float16)p[j * es];
  return b;
}

// ---------------------------------------------------------------------------
// Kernel A: bulk f32 -> f16 conversion of q,k,v (one bandwidth pass).
// grid = (2M/(256*8), 1, 3)
// ---------------------------------------------------------------------------
__global__ __launch_bounds__(256) void cvt_qkv_kernel(
    const float* __restrict__ q, const float* __restrict__ k,
    const float* __restrict__ v, _Float16* __restrict__ qh,
    _Float16* __restrict__ kh, _Float16* __restrict__ vh) {
  const int t = blockIdx.z;
  const float* src = (t == 0) ? q : ((t == 1) ? k : v);
  _Float16* dst    = (t == 0) ? qh : ((t == 1) ? kh : vh);
  size_t base = ((size_t)blockIdx.x * 256 + threadIdx.x) * 8;
#pragma unroll
  for (int j = 0; j < 8; ++j) dst[base + j] = (_Float16)src[base + j];
}

// ---------------------------------------------------------------------------
// Kernel B: projection weights [H,512,64] f32 -> Wt[t][h][e][k] f16
// (k contiguous, so proj B-fragments are unit-stride). grid = (512, 8, 3)x64
// ---------------------------------------------------------------------------
__global__ __launch_bounds__(64) void cvt_w_kernel(
    const float* __restrict__ wq, const float* __restrict__ wk,
    const float* __restrict__ wv, _Float16* __restrict__ wt) {
  const int kd = blockIdx.x;   // 0..511
  const int h  = blockIdx.y;   // 0..7
  const int t  = blockIdx.z;   // 0..2
  const int e  = threadIdx.x;  // 0..63
  const float* W = (t == 0) ? wq : ((t == 1) ? wk : wv);
  float val = W[((size_t)h * DMODEL + kd) * DK + e];
  wt[(((size_t)t * NHEAD + h) * DK + e) * DMODEL + kd] = (_Float16)val;
}

// ---------------------------------------------------------------------------
// Kernel 0: per-head spreads -> 1/(2*sigma^2)
// ---------------------------------------------------------------------------
__global__ __launch_bounds__(32) void spreads_kernel(
    const float* __restrict__ sw, float* __restrict__ inv2s2) {
  int h = threadIdx.x;
  if (h < NHEAD) {
    float wv[NHEAD];
    float mx = -1e30f;
#pragma unroll
    for (int j = 0; j < NHEAD; ++j) { wv[j] = sw[h * NHEAD + j]; mx = fmaxf(mx, wv[j]); }
    float s = 0.0f;
#pragma unroll
    for (int j = 0; j < NHEAD; ++j) { wv[j] = __expf(wv[j] - mx); s += wv[j]; }
    float spread = 0.0f;
#pragma unroll
    for (int j = 0; j < NHEAD; ++j) {
      float raw  = __powf(20.0f, (float)j * (1.0f / 7.0f));      // logspace base 20
      float base = 1.0f + 5.0f * (raw - 1.0f) / 19.0f;           // [1,6]
      spread += (wv[j] / s) * base;
    }
    inv2s2[h] = 1.0f / (2.0f * spread * spread);
  }
}

// ---------------------------------------------------------------------------
// Kernel 1: per-head projection + fused LayerNorm.
// Q,K -> f16 [B,H,N,DK];  V -> f16 [B,H,DK,N] (transposed for P*V).
// One wave computes a 16x64 output tile. grid = (4096/16, H, 3)
// ---------------------------------------------------------------------------
__global__ __launch_bounds__(32) void proj_ln_kernel(
    const _Float16* __restrict__ qh, const _Float16* __restrict__ kh,
    const _Float16* __restrict__ vh, const _Float16* __restrict__ wt,
    const float* __restrict__ qg, const float* __restrict__ qb,
    const float* __restrict__ kg, const float* __restrict__ kb2,
    const float* __restrict__ vg, const float* __restrict__ vb,
    _Float16* __restrict__ Qo, _Float16* __restrict__ Ko,
    _Float16* __restrict__ Vt) {
  const int lane  = threadIdx.x;
  const int rt    = blockIdx.x;          // 16-row tile over 4096 flattened rows
  const int h     = blockIdx.y;
  const int t     = blockIdx.z;
  const int l15   = lane & 15;
  const int half_ = (lane >> 4) & 1;
  const int ko    = half_ * 16;

  const _Float16* X; const float* g; const float* bb;
  if (t == 0)      { X = qh; g = qg; bb = qb;  }
  else if (t == 1) { X = kh; g = kg; bb = kb2; }
  else             { X = vh; g = vg; bb = vb;  }
  const _Float16* Wth = wt + (((size_t)t * NHEAD + h) * DK) * DMODEL;

  const int row = rt * 16 + l15;                    // A-fragment row

  v8f acc[4];
#pragma unroll
  for (int f = 0; f < 4; ++f) acc[f] = zero8();

  for (int kk = 0; kk < DMODEL; kk += 32) {
    v16h a = load_a_f16(X + (size_t)row * DMODEL + kk, lane);
#pragma unroll
    for (int f = 0; f < 4; ++f) {
      // B(k,c) = Wt[c][k]  -> unit stride in k
      v16h bfr = load_b_f16(Wth + (size_t)(f * 16 + l15) * DMODEL + kk + ko, 1);
      acc[f] = wmma_f16(a, bfr, acc[f]);
    }
  }

  // Fused LayerNorm over DK=64: row m lives in VGPR i across 16 lanes of a half.
#pragma unroll
  for (int i = 0; i < 8; ++i) {
    float s1 = 0.0f, s2 = 0.0f;
#pragma unroll
    for (int f = 0; f < 4; ++f) { float x = acc[f][i]; s1 += x; s2 += x * x; }
#pragma unroll
    for (int off = 1; off < 16; off <<= 1) {
      s1 += __shfl_xor(s1, off, 32);
      s2 += __shfl_xor(s2, off, 32);
    }
    float mean = s1 * (1.0f / 64.0f);
    float var  = s2 * (1.0f / 64.0f) - mean * mean;
    float rstd = rsqrtf(var + 1e-5f);

    int grow = rt * 16 + i + 8 * half_;             // global flat row
    int bI = grow >> 10, n = grow & (NSEQ - 1);
#pragma unroll
    for (int f = 0; f < 4; ++f) {
      int c = f * 16 + l15;
      _Float16 y = (_Float16)((acc[f][i] - mean) * rstd * g[c] + bb[c]);
      if (t == 2) {
        // transposed: Vt[b][h][c][n]
        Vt[(((size_t)bI * NHEAD + h) * DK + c) * NSEQ + n] = y;
      } else {
        _Float16* O = (t == 0) ? Qo : Ko;
        O[(((size_t)bI * NHEAD + h) * NSEQ + n) * DK + c] = y;
      }
    }
  }
}

// ---------------------------------------------------------------------------
// Kernel 2: flash-style geometric attention.
// One wave handles 16 query rows of one (b,h); loops 32-key blocks:
//   S = Q K^T (WMMA), logits = S/8 * clamp(exp(-d2/(2s^2)),.1,.9),
//   online softmax, P transposed via LDS, O += P V (WMMA, Vt unit-stride).
// grid = (N/16, H, B)
// ---------------------------------------------------------------------------
__global__ __launch_bounds__(32) void attn_kernel(
    const _Float16* __restrict__ Qf, const _Float16* __restrict__ Kf,
    const _Float16* __restrict__ Vt, const float* __restrict__ coords,
    const float* __restrict__ inv2s2, _Float16* __restrict__ Xo) {
  __shared__ _Float16 Plds[16 * 32];

  const int lane  = threadIdx.x;
  const int qt    = blockIdx.x;
  const int h     = blockIdx.y;
  const int b     = blockIdx.z;
  const int l15   = lane & 15;
  const int half_ = (lane >> 4) & 1;
  const int ko    = half_ * 16;
  const int kb    = half_ * 16;

  const _Float16* Qp  = Qf + (((size_t)b * NHEAD + h) * NSEQ) * DK;
  const _Float16* Kp  = Kf + (((size_t)b * NHEAD + h) * NSEQ) * DK;
  const _Float16* Vtp = Vt + (((size_t)b * NHEAD + h) * DK) * NSEQ;
  const float*    cb  = coords + (size_t)b * NSEQ * 3;

  const int qrow = qt * 16 + l15;
  v16h aQ0 = load_a_f16(Qp + (size_t)qrow * DK + 0,  lane);
  v16h aQ1 = load_a_f16(Qp + (size_t)qrow * DK + 32, lane);

  float qx[8], qy[8], qz[8];
#pragma unroll
  for (int i = 0; i < 8; ++i) {
    int m = qt * 16 + i + 8 * half_;
    qx[i] = cb[m * 3 + 0]; qy[i] = cb[m * 3 + 1]; qz[i] = cb[m * 3 + 2];
  }

  const float inv   = inv2s2[h];
  const float scale = 0.125f;  // 1/sqrt(64)

  float rmax[8], rsum[8];
  v8f   Oacc[4];
#pragma unroll
  for (int i = 0; i < 8; ++i) { rmax[i] = -1e30f; rsum[i] = 0.0f; }
#pragma unroll
  for (int f = 0; f < 4; ++f) Oacc[f] = zero8();

  for (int k0 = 0; k0 < NSEQ; k0 += 32) {
    // Prefetch next key block (K rows + V columns) -> global_prefetch_b8
    if (k0 + 32 < NSEQ) {
      __builtin_prefetch(Kp + (size_t)(k0 + 32 + lane) * DK, 0, 0);
      __builtin_prefetch(Vtp + (size_t)(lane << 1) * NSEQ + k0 + 32, 0, 0);
    }

    // ---- S = Q K^T for 16 queries x 32 keys (2 N-tiles x 2 K-chunks) ----
    v8f S0 = zero8(), S1 = zero8();
    {
      const _Float16* kp0 = Kp + (size_t)(k0 + l15) * DK;
      S0 = wmma_f16(aQ0, load_b_f16(kp0 + 0  + ko, 1), S0);
      S0 = wmma_f16(aQ1, load_b_f16(kp0 + 32 + ko, 1), S0);
      const _Float16* kp1 = Kp + (size_t)(k0 + 16 + l15) * DK;
      S1 = wmma_f16(aQ0, load_b_f16(kp1 + 0  + ko, 1), S1);
      S1 = wmma_f16(aQ1, load_b_f16(kp1 + 32 + ko, 1), S1);
    }

    // ---- RBF modulation + online softmax ----
    int n0 = k0 + l15, n1 = n0 + 16;
    float k0x = cb[n0 * 3 + 0], k0y = cb[n0 * 3 + 1], k0z = cb[n0 * 3 + 2];
    float k1x = cb[n1 * 3 + 0], k1y = cb[n1 * 3 + 1], k1z = cb[n1 * 3 + 2];

#pragma unroll
    for (int i = 0; i < 8; ++i) {
      float dx = qx[i] - k0x, dy = qy[i] - k0y, dz = qz[i] - k0z;
      float r0 = __expf(-(dx * dx + dy * dy + dz * dz) * inv);
      r0 = fminf(fmaxf(r0, 0.1f), 0.9f);
      dx = qx[i] - k1x; dy = qy[i] - k1y; dz = qz[i] - k1z;
      float r1 = __expf(-(dx * dx + dy * dy + dz * dz) * inv);
      r1 = fminf(fmaxf(r1, 0.1f), 0.9f);

      float l0 = S0[i] * scale * r0;
      float l1 = S1[i] * scale * r1;

      float mx = fmaxf(l0, l1);
#pragma unroll
      for (int off = 1; off < 16; off <<= 1) mx = fmaxf(mx, __shfl_xor(mx, off, 32));
      float nm   = fmaxf(rmax[i], mx);
      float corr = __expf(rmax[i] - nm);
      rmax[i] = nm;

      float p0 = __expf(l0 - nm), p1 = __expf(l1 - nm);
      float ps = p0 + p1;
#pragma unroll
      for (int off = 1; off < 16; off <<= 1) ps += __shfl_xor(ps, off, 32);
      rsum[i] = rsum[i] * corr + ps;
#pragma unroll
      for (int f = 0; f < 4; ++f) Oacc[f][i] *= corr;

      int m = i + 8 * half_;
      Plds[m * 32 + l15]      = (_Float16)p0;
      Plds[m * 32 + l15 + 16] = (_Float16)p1;
    }
    __syncthreads();

    // transpose P (C-layout) -> A-fragment via LDS
    v16h aP;
#pragma unroll
    for (int j = 0; j < 8; ++j) {
      aP[j]     = Plds[l15 * 32 + kb + j];
      aP[8 + j] = Plds[l15 * 32 + kb + 8 + j];
    }
    __syncthreads();

    // ---- O += P * V  (B(k,c) = Vt[c][k0+k], unit stride in k) ----
#pragma unroll
    for (int f = 0; f < 4; ++f) {
      v16h bv = load_b_f16(Vtp + (size_t)(f * 16 + l15) * NSEQ + k0 + ko, 1);
      Oacc[f] = wmma_f16(aP, bv, Oacc[f]);
    }
  }

  // ---- normalize and scatter to X layout [B,N, e*H + h] as f16 ----
#pragma unroll
  for (int i = 0; i < 8; ++i) {
    float rinv = 1.0f / rsum[i];
    int n = qt * 16 + i + 8 * half_;
    _Float16* xr = Xo + ((size_t)b * NSEQ + n) * DMODEL;
#pragma unroll
    for (int f = 0; f < 4; ++f) {
      int e = f * 16 + l15;
      xr[e * NHEAD + h] = (_Float16)(Oacc[f][i] * rinv);
    }
  }
}

// ---------------------------------------------------------------------------
// Kernel 3: Y = X @ W^T + bias, f32 out. grid = (4096/16, 512/16)
// W^T access is already unit-stride (B(c,o) = W[o][c]).
// ---------------------------------------------------------------------------
__global__ __launch_bounds__(32) void out_gemm_kernel(
    const _Float16* __restrict__ X, const float* __restrict__ W,
    const float* __restrict__ bias, float* __restrict__ Y) {
  const int lane  = threadIdx.x;
  const int rt    = blockIdx.x;
  const int ct    = blockIdx.y;
  const int l15   = lane & 15;
  const int half_ = (lane >> 4) & 1;
  const int ko    = half_ * 16;
  const int row   = rt * 16 + l15;
  const int o     = ct * 16 + l15;

  v8f acc = zero8();
  for (int kk = 0; kk < DMODEL; kk += 32) {
    v16h a   = load_a_f16(X + (size_t)row * DMODEL + kk, lane);
    v16h bfr = load_b_f32(W + (size_t)o * DMODEL + kk + ko, 1);
    acc = wmma_f16(a, bfr, acc);
  }

  float bv = bias[o];
#pragma unroll
  for (int i = 0; i < 8; ++i) {
    int r = rt * 16 + i + 8 * half_;
    Y[(size_t)r * DMODEL + ct * 16 + l15] = acc[i] + bv;
  }
}

// ---------------------------------------------------------------------------
extern "C" void kernel_launch(void* const* d_in, const int* in_sizes, int n_in,
                              void* d_out, int out_size, void* d_ws, size_t ws_size,
                              hipStream_t stream) {
  (void)in_sizes; (void)n_in; (void)out_size; (void)ws_size;
  const float* q      = (const float*)d_in[0];
  const float* k      = (const float*)d_in[1];
  const float* v      = (const float*)d_in[2];
  const float* coords = (const float*)d_in[3];
  const float* wq     = (const float*)d_in[4];
  const float* wk     = (const float*)d_in[5];
  const float* wv     = (const float*)d_in[6];
  const float* sw     = (const float*)d_in[7];
  const float* qg     = (const float*)d_in[8];
  const float* qb     = (const float*)d_in[9];
  const float* kg     = (const float*)d_in[10];
  const float* kb     = (const float*)d_in[11];
  const float* vg     = (const float*)d_in[12];
  const float* vb     = (const float*)d_in[13];
  const float* ow     = (const float*)d_in[14];
  const float* ob     = (const float*)d_in[15];
  float* out = (float*)d_out;

  char* ws = (char*)d_ws;
  float*    inv2s2 = (float*)ws;                              // 64 B slot
  const size_t perH = (size_t)BATCH * NHEAD * NSEQ * DK;      // 2M halves
  const size_t perX = (size_t)BATCH * NSEQ * DMODEL;          // 2M halves
  _Float16* Qf = (_Float16*)(ws + 64);
  _Float16* Kf = Qf + perH;
  _Float16* Vt = Kf + perH;        // [B,H,DK,N]
  _Float16* Xf = Vt + perH;        // attention output, [B,N,512]
  _Float16* qh = Xf + perX;        // f16 copies of q,k,v
  _Float16* kh = qh + perX;
  _Float16* vh = kh + perX;
  _Float16* wt = vh + perX;        // [3,H,DK,DMODEL]

  cvt_qkv_kernel<<<dim3(perX / (256 * 8), 1, 3), dim3(256), 0, stream>>>(
      q, k, v, qh, kh, vh);
  cvt_w_kernel<<<dim3(DMODEL, NHEAD, 3), dim3(64), 0, stream>>>(wq, wk, wv, wt);
  spreads_kernel<<<dim3(1), dim3(32), 0, stream>>>(sw, inv2s2);
  proj_ln_kernel<<<dim3((BATCH * NSEQ) / 16, NHEAD, 3), dim3(32), 0, stream>>>(
      qh, kh, vh, wt, qg, qb, kg, kb, vg, vb, Qf, Kf, Vt);
  attn_kernel<<<dim3(NSEQ / 16, NHEAD, BATCH), dim3(32), 0, stream>>>(
      Qf, Kf, Vt, coords, inv2s2, Xf);
  out_gemm_kernel<<<dim3((BATCH * NSEQ) / 16, DMODEL / 16), dim3(32), 0, stream>>>(
      Xf, ow, ob, out);
}
